// HyperIntelligentAI_60868276519721
// MI455X (gfx1250) — compile-verified, matrix operand order checked by reading, not run on hardware
//
#include <hip/hip_runtime.h>
#include <hip/hip_bf16.h>

// ---------------- problem constants ----------------
#define NE 16      // experts
#define DM 1024    // d_model
#define FF 2048    // d_ff
#define NT 4096    // tokens (B*N)
#define LN_EPS 1e-5f
#define MT 32      // tokens per FFN block tile

// ---------------- vector types ----------------
typedef __bf16 bf16x8  __attribute__((ext_vector_type(8)));
typedef __bf16 bf16x16 __attribute__((ext_vector_type(16)));
typedef float  floatx4 __attribute__((ext_vector_type(4)));
typedef float  floatx8 __attribute__((ext_vector_type(8)));

// Fragment loaders. Per-lane K pattern (ISA 7.12.2, 16-bit A/B 16x32):
// lanes 0-15: K = [k0..k0+7, k0+16..k0+23]; lanes 16-31: +8  -> base = k0+8*half.
static __device__ inline bf16x16 bf_frag(const __bf16* row, int k0, int half) {
    const bf16x8* p = (const bf16x8*)(row + k0 + 8 * half);
    const bf16x8* q = (const bf16x8*)(row + k0 + 16 + 8 * half);
    bf16x8 lo = *p, hi = *q;
    return __builtin_shufflevector(lo, hi, 0,1,2,3,4,5,6,7,8,9,10,11,12,13,14,15);
}
// fp32-source fallback (on-the-fly convert; used only when ws too small)
static __device__ inline bf16x16 bf_frag(const float* row, int k0, int half) {
    const floatx4* p = (const floatx4*)(row + k0 + 8 * half);
    const floatx4* q = (const floatx4*)(row + k0 + 16 + 8 * half);
    floatx4 a0 = p[0], a1 = p[1], b0 = q[0], b1 = q[1];
    bf16x16 r;
#pragma unroll
    for (int i = 0; i < 4; ++i) {
        r[i]      = (__bf16)a0[i];
        r[4 + i]  = (__bf16)a1[i];
        r[8 + i]  = (__bf16)b0[i];
        r[12 + i] = (__bf16)b1[i];
    }
    return r;
}

// ---------------- kernel 0: zero output + expert counters ----------------
__global__ void moe_zero(float* __restrict__ out, int* __restrict__ cnt) {
    int i = blockIdx.x * blockDim.x + threadIdx.x;
    floatx4 z = {0.f, 0.f, 0.f, 0.f};
    ((floatx4*)out)[i] = z;                         // 4096 blocks * 256 thr * 4 = NT*DM
    if (blockIdx.x == 0 && threadIdx.x < NE) cnt[threadIdx.x] = 0;
}

// ---------------- kernel 0b: fp32 weights -> bf16 (one-time, 8 elem/thread) --
__global__ void moe_convw(const float* __restrict__ s, __bf16* __restrict__ d) {
    size_t i = ((size_t)blockIdx.x * blockDim.x + threadIdx.x) * 8;
    floatx4 a = *(const floatx4*)(s + i);
    floatx4 b = *(const floatx4*)(s + i + 4);
    bf16x8 o;
#pragma unroll
    for (int j = 0; j < 4; ++j) { o[j] = (__bf16)a[j]; o[4 + j] = (__bf16)b[j]; }
    *(bf16x8*)(d + i) = o;
}

// ---------------- kernel 1: router (logits, softmax, top-2, gather lists) ---
__global__ void moe_router(const float* __restrict__ x,
                           const float* __restrict__ rw,
                           const float* __restrict__ rb,
                           int* __restrict__ cnt,
                           int* __restrict__ idxL,
                           float* __restrict__ cfL) {
    int t = blockIdx.x * blockDim.x + threadIdx.x;
    if (t >= NT) return;
    float acc[NE];
#pragma unroll
    for (int e = 0; e < NE; ++e) acc[e] = rb[e];
    const float* xr = x + (size_t)t * DM;
    for (int d = 0; d < DM; ++d) {
        float xv = xr[d];
#pragma unroll
        for (int e = 0; e < NE; ++e) acc[e] += xv * rw[e * DM + d];
    }
    float m = acc[0];
#pragma unroll
    for (int e = 1; e < NE; ++e) m = fmaxf(m, acc[e]);
    float s = 0.f;
#pragma unroll
    for (int e = 0; e < NE; ++e) { acc[e] = expf(acc[e] - m); s += acc[e]; }
    float inv = 1.f / s;
#pragma unroll
    for (int e = 0; e < NE; ++e) acc[e] *= inv;
    // top-2, strict-greater => lower index wins ties (lax.top_k semantics)
    int e0, e1; float w0, w1;
    if (acc[1] > acc[0]) { e0 = 1; w0 = acc[1]; e1 = 0; w1 = acc[0]; }
    else                 { e0 = 0; w0 = acc[0]; e1 = 1; w1 = acc[1]; }
#pragma unroll
    for (int e = 2; e < NE; ++e) {
        float p = acc[e];
        if (p > w0)      { e1 = e0; w1 = w0; e0 = e; w0 = p; }
        else if (p > w1) { e1 = e;  w1 = p; }
    }
    float rs = 1.f / (w0 + w1);
    int p0 = atomicAdd(&cnt[e0], 1);
    idxL[e0 * NT + p0] = t;  cfL[e0 * NT + p0] = w0 * rs;
    int p1 = atomicAdd(&cnt[e1], 1);
    idxL[e1 * NT + p1] = t;  cfL[e1 * NT + p1] = w1 * rs;
}

// ---------------- kernel 2: fused expert FFN + LayerNorm + scatter ----------
// LDS layout (dynamic), 32-token tile:
#define HH_OFF 0            // 32*2048*2 = 131072  (h tile, bf16)
#define XS_OFF 131072       // 32*1024*2 =  65536  (gathered x, bf16; dead after ph1)
#define YS_OFF 131072       // 32*1024*4 = 131072  (y tile fp32; overlays dead xs)
#define TK_OFF 262144       // 32*4
#define CF_OFF 262272       // 32*4
#define SMEM_BYTES 262400

template <typename WT>
__global__ void moe_ffn(const float* __restrict__ x,
                        const WT* __restrict__ w1,
                        const WT* __restrict__ w2,
                        const WT* __restrict__ w3,
                        const float* __restrict__ lng,
                        const float* __restrict__ lnb,
                        const int* __restrict__ cnt,
                        const int* __restrict__ idxL,
                        const float* __restrict__ cfL,
                        float* __restrict__ out) {
    extern __shared__ char smem[];
    __bf16* hh   = (__bf16*)(smem + HH_OFF);
    __bf16* xs   = (__bf16*)(smem + XS_OFF);
    float*  ys   = (float*)(smem + YS_OFF);
    int*    toks = (int*)(smem + TK_OFF);
    float*  cfs  = (float*)(smem + CF_OFF);

    const int e     = blockIdx.y;
    const int tile  = blockIdx.x;
    const int count = cnt[e];
    if (tile * MT >= count) return;              // uniform early-exit (EXEC stays full)

    const int tid  = threadIdx.x;
    const int wave = tid >> 5;
    const int lane = tid & 31;
    const int half = lane >> 4;
    const int mn   = lane & 15;                  // A row (M) == B col (N) lane index

    if (tid < MT) {
        int g = tile * MT + tid;
        if (g < count) { toks[tid] = idxL[e * NT + g]; cfs[tid] = cfL[e * NT + g]; }
        else           { toks[tid] = 0;                cfs[tid] = 0.f; }
    }
    __syncthreads();

    // gather x tile -> LDS as bf16 (coalesced)
    for (int i = tid; i < MT * DM; i += 256) {
        int r = i >> 10, d = i & (DM - 1);
        xs[i] = (__bf16)x[(size_t)toks[r] * DM + d];
    }
    __syncthreads();

    // ---- phase 1: h = silu(x@W1^T) * (x@W3^T), 32 x 2048 ----
    // Each B fragment (w1/w3) feeds 2 WMMAs (two 16-token M-tiles).
    const __bf16* xrow0 = xs + mn * DM;
    const __bf16* xrow1 = xs + (16 + mn) * DM;
    for (int nt = wave; nt < FF / 16; nt += 8) {
        const int nBase = nt * 16;
        const WT* r1 = w1 + ((size_t)e * FF + nBase + mn) * DM;
        const WT* r3 = w3 + ((size_t)e * FF + nBase + mn) * DM;
        floatx8 a10 = {0,0,0,0,0,0,0,0}, a11 = a10, a30 = a10, a31 = a10;
        for (int k0 = 0; k0 < DM; k0 += 32) {
            __builtin_prefetch(r1 + k0 + 64, 0, 1);
            __builtin_prefetch(r3 + k0 + 64, 0, 1);
            bf16x16 af0 = bf_frag(xrow0, k0, half);
            bf16x16 af1 = bf_frag(xrow1, k0, half);
            bf16x16 b1  = bf_frag(r1, k0, half);
            bf16x16 b3  = bf_frag(r3, k0, half);
            a10 = __builtin_amdgcn_wmma_f32_16x16x32_bf16(false, af0, false, b1,
                                                          (short)0, a10, false, false);
            a11 = __builtin_amdgcn_wmma_f32_16x16x32_bf16(false, af1, false, b1,
                                                          (short)0, a11, false, false);
            a30 = __builtin_amdgcn_wmma_f32_16x16x32_bf16(false, af0, false, b3,
                                                          (short)0, a30, false, false);
            a31 = __builtin_amdgcn_wmma_f32_16x16x32_bf16(false, af1, false, b3,
                                                          (short)0, a31, false, false);
        }
#pragma unroll
        for (int r = 0; r < 8; ++r) {            // C layout: row = r + 8*half
            int row0 = r + 8 * half;
            float s1 = a10[r], h0 = (s1 / (1.f + __expf(-s1))) * a30[r];
            hh[row0 * FF + nBase + mn] = (__bf16)h0;
            float s2 = a11[r], h1 = (s2 / (1.f + __expf(-s2))) * a31[r];
            hh[(16 + row0) * FF + nBase + mn] = (__bf16)h1;
        }
    }
    __syncthreads();

    // ---- phase 2: y = h @ W2^T, 32 x 1024 ----
    const __bf16* hrow0 = hh + mn * FF;
    const __bf16* hrow1 = hh + (16 + mn) * FF;
    for (int dt = wave; dt < DM / 16; dt += 8) {
        const int dBase = dt * 16;
        const WT* r2 = w2 + ((size_t)e * DM + dBase + mn) * FF;
        floatx8 ay0 = {0,0,0,0,0,0,0,0}, ay1 = ay0;
        for (int f0 = 0; f0 < FF; f0 += 32) {
            __builtin_prefetch(r2 + f0 + 64, 0, 1);
            bf16x16 af0 = bf_frag(hrow0, f0, half);
            bf16x16 af1 = bf_frag(hrow1, f0, half);
            bf16x16 bf  = bf_frag(r2, f0, half);
            ay0 = __builtin_amdgcn_wmma_f32_16x16x32_bf16(false, af0, false, bf,
                                                          (short)0, ay0, false, false);
            ay1 = __builtin_amdgcn_wmma_f32_16x16x32_bf16(false, af1, false, bf,
                                                          (short)0, ay1, false, false);
        }
#pragma unroll
        for (int r = 0; r < 8; ++r) {
            int row0 = r + 8 * half;
            ys[row0 * DM + dBase + mn]        = ay0[r];
            ys[(16 + row0) * DM + dBase + mn] = ay1[r];
        }
    }
    __syncthreads();

    // ---- phase 3: LayerNorm over D, scale by routing coeff, scatter-add ----
    const int tok = tid >> 3;    // 0..31 (token slot); 8 threads per token
    const int grp = tid & 7;
    const float* yr = ys + tok * DM;
    float s = 0.f, sq = 0.f;
    for (int j = grp; j < DM; j += 8) { float v = yr[j]; s += v; sq += v * v; }
#pragma unroll
    for (int off = 4; off > 0; off >>= 1) {      // stays within 8-lane groups of wave32
        s  += __shfl_xor(s,  off, 32);
        sq += __shfl_xor(sq, off, 32);
    }
    const float mean = s * (1.f / DM);
    const float var  = sq * (1.f / DM) - mean * mean;   // biased (torch LayerNorm)
    const float rstd = rsqrtf(var + LN_EPS);
    const float cf   = cfs[tok];
    float* orow = out + (size_t)toks[tok] * DM;
    const float* g = lng + (size_t)e * DM;
    const float* b = lnb + (size_t)e * DM;
    for (int j = grp; j < DM; j += 8) {
        float yn = (yr[j] - mean) * rstd * g[j] + b[j];
        atomicAdd(&orow[j], cf * yn);            // exactly 2 adds/elem: commutative
    }
}

// ---------------- host-side launch ----------------
extern "C" void kernel_launch(void* const* d_in, const int* in_sizes, int n_in,
                              void* d_out, int out_size, void* d_ws, size_t ws_size,
                              hipStream_t stream) {
    const float* x   = (const float*)d_in[0];
    const float* rw  = (const float*)d_in[1];
    const float* rb  = (const float*)d_in[2];
    const float* w1  = (const float*)d_in[3];
    const float* w2  = (const float*)d_in[4];
    const float* w3  = (const float*)d_in[5];
    const float* lng = (const float*)d_in[6];
    const float* lnb = (const float*)d_in[7];
    float* out = (float*)d_out;

    // ws: [cnt 16i][pad->256B][idxL 16*4096 i32][cfL 16*4096 f32][opt: bf16 weights]
    int*   cnt  = (int*)d_ws;
    int*   idxL = (int*)((char*)d_ws + 256);
    float* cfL  = (float*)((char*)d_ws + 256 + (size_t)NE * NT * sizeof(int));
    const size_t base   = 256 + 2 * (size_t)NE * NT * 4;            // 524544
    const size_t wElems = (size_t)NE * FF * DM;                     // 32M per tensor
    const bool   pre    = ws_size >= base + 3 * wElems * sizeof(__bf16);

    moe_zero<<<dim3(NT * DM / (256 * 4)), dim3(256), 0, stream>>>(out, cnt);
    moe_router<<<dim3(NT / 256), dim3(256), 0, stream>>>(x, rw, rb, cnt, idxL, cfL);

    if (pre) {
        __bf16* w1b = (__bf16*)((char*)d_ws + base);
        __bf16* w2b = w1b + wElems;
        __bf16* w3b = w2b + wElems;
        const int cblk = (int)(wElems / (256 * 8));                 // 16384 blocks
        moe_convw<<<dim3(cblk), dim3(256), 0, stream>>>(w1, w1b);
        moe_convw<<<dim3(cblk), dim3(256), 0, stream>>>(w2, w2b);
        moe_convw<<<dim3(cblk), dim3(256), 0, stream>>>(w3, w3b);
        moe_ffn<__bf16><<<dim3(NT / MT, NE), dim3(256), SMEM_BYTES, stream>>>(
            x, w1b, w2b, w3b, lng, lnb, cnt, idxL, cfL, out);
    } else {
        moe_ffn<float><<<dim3(NT / MT, NE), dim3(256), SMEM_BYTES, stream>>>(
            x, w1, w2, w3, lng, lnb, cnt, idxL, cfL, out);
    }
}
